// CausalSelfAttention_73358041416064
// MI455X (gfx1250) — matrix operand order repaired and checked by reading, not compiled, not run on hardware
//
#include <hip/hip_runtime.h>
#include <hip/hip_bf16.h>

// ---------------------------------------------------------------------------
// Causal self-attention forward for MI455X (gfx1250, wave32, WMMA).
// All matmuls on v_wmma_f32_16x16x32_bf16 (f32 accum). GEMMs use
// double-buffered LDS fed by GLOBAL_LOAD_ASYNC_TO_LDS_B128 (ASYNCcnt path).
// ---------------------------------------------------------------------------

typedef __bf16 bf16;
typedef __attribute__((ext_vector_type(16))) __bf16 v16bf;
typedef __attribute__((ext_vector_type(8)))  __bf16 v8bf;
typedef __attribute__((ext_vector_type(4)))  __bf16 v4bf;
typedef __attribute__((ext_vector_type(8)))  float  v8f;
typedef __attribute__((ext_vector_type(4)))  int    v4i;

static constexpr int Bn  = 2;
static constexpr int Sn  = 2048;
static constexpr int Dn  = 2048;
static constexpr int Hn  = 16;
static constexpr int HDn = 128;
static constexpr float SCALE = 0.022097086912079608f; // 1/sqrt(2048)

#define WMMA_BF16(a, b, c) \
  __builtin_amdgcn_wmma_f32_16x16x32_bf16(false, (a), false, (b), (short)0, (c), false, false)

#if __has_builtin(__builtin_amdgcn_global_load_async_to_lds_b128)
#define ATHENA_ASYNC_LDS 1
#else
#define ATHENA_ASYNC_LDS 0
#endif

// 16B global -> LDS copy. Async DMA (no VGPR staging) when available.
__device__ __forceinline__ void copy16_g2l(const bf16* __restrict__ gsrc,
                                           bf16* __restrict__ ldst) {
#if ATHENA_ASYNC_LDS
  __builtin_amdgcn_global_load_async_to_lds_b128(
      (__attribute__((address_space(1))) v4i*)(void*)gsrc,
      (__attribute__((address_space(3))) v4i*)(void*)ldst, 0, 0);
#else
  *(v8bf*)ldst = *(const v8bf*)gsrc;
#endif
}

__device__ __forceinline__ void wait_async_all() {
#if ATHENA_ASYNC_LDS
#if __has_builtin(__builtin_amdgcn_s_wait_asynccnt)
  __builtin_amdgcn_s_wait_asynccnt(0);
#else
  asm volatile("s_wait_asynccnt 0x0" ::: "memory");
#endif
#endif
}

__device__ __forceinline__ v8f vzero8() {
  v8f z;
#pragma unroll
  for (int i = 0; i < 8; ++i) z[i] = 0.0f;
  return z;
}

// A fragment (16x32 bf16, M x K), row-major source with `stride` elements/row.
// Lane l: row = l&15, half = l>>4. elems 0..7 = K[8h..8h+7], 8..15 = K[16+8h..].
__device__ __forceinline__ v16bf frag_a(const bf16* __restrict__ p0, int stride) {
  const int lane = threadIdx.x & 31;
  const int m = lane & 15, h = lane >> 4;
  const bf16* p = p0 + m * stride + h * 8;
  v8bf lo = *(const v8bf*)(p);
  v8bf hi = *(const v8bf*)(p + 16);
  v16bf r;
#pragma unroll
  for (int i = 0; i < 8; ++i) { r[i] = lo[i]; r[8 + i] = hi[i]; }
  return r;
}

// B fragment (32x16 bf16, K x N) sourced from a row-major [N][K] tile (B^T),
// so lane (col n = l&15) reads K[8h..8h+7] and K[16+8h..16+8h+7].
__device__ __forceinline__ v16bf frag_b(const bf16* __restrict__ p0, int stride) {
  const int lane = threadIdx.x & 31;
  const int n = lane & 15, h = lane >> 4;
  const bf16* p = p0 + n * stride + h * 8;
  v8bf lo = *(const v8bf*)(p);
  v8bf hi = *(const v8bf*)(p + 16);
  v16bf r;
#pragma unroll
  for (int i = 0; i < 8; ++i) { r[i] = lo[i]; r[8 + i] = hi[i]; }
  return r;
}

// ---------------------------------------------------------------------------
// fp32 -> bf16 conversion (4-wide)
// ---------------------------------------------------------------------------
__global__ void __launch_bounds__(256)
cvt_f32_bf16(const float* __restrict__ s, bf16* __restrict__ d, int n4) {
  int i = blockIdx.x * blockDim.x + threadIdx.x;
  const int stride = gridDim.x * blockDim.x;
  for (; i < n4; i += stride) {
    float4 v = ((const float4*)s)[i];
    v4bf o;
    o[0] = (bf16)v.x; o[1] = (bf16)v.y; o[2] = (bf16)v.z; o[3] = (bf16)v.w;
    ((v4bf*)d)[i] = o;
  }
}

// ---------------------------------------------------------------------------
// C[M,N] = A[M,K] @ Bw[N,K]^T   (bf16 in, f32 accumulate, OutT out)
// Block: 256 threads = 8 waves. Block tile 128x128, K-step 32.
// Double-buffered LDS; async global->LDS; ONE barrier per K-step.
// Wave grid 2x4: each wave owns 64(M) x 32(N) = 4x2 tiles of 16x16.
// ---------------------------------------------------------------------------
template <typename OutT>
__global__ void __launch_bounds__(256)
gemm_bf16_wmma(const bf16* __restrict__ A, const bf16* __restrict__ Bw,
               OutT* __restrict__ C, int M, int N, int K,
               int lda, int ldb, int ldc) {
  __shared__ bf16 As[2][128][40];  // 32 K + pad 8 (80B stride, 16B aligned)
  __shared__ bf16 Bs[2][128][40];

  const int tid = threadIdx.x;
  const int lane = tid & 31, wave = tid >> 5;
  const int wm = wave >> 2, wn = wave & 3;
  const int m0 = blockIdx.x * 128, n0 = blockIdx.y * 128;
  const int h = lane >> 4, nlo = lane & 15;

  // This thread's two (row, chunk) slots: 128 rows x 4 x 16B chunks / matrix.
  const int r0 = tid >> 2, c0 = (tid & 3) << 3;          // slot 0
  const int r1 = (tid + 256) >> 2, c1 = c0;              // slot 1

  v8f acc[4][2];
#pragma unroll
  for (int mi = 0; mi < 4; ++mi)
#pragma unroll
    for (int ni = 0; ni < 2; ++ni) acc[mi][ni] = vzero8();

  // prologue: stage k0 = 0 into buffer 0
  {
    copy16_g2l(A + (size_t)(m0 + r0) * lda + c0, &As[0][r0][c0]);
    copy16_g2l(Bw + (size_t)(n0 + r0) * ldb + c0, &Bs[0][r0][c0]);
    copy16_g2l(A + (size_t)(m0 + r1) * lda + c1, &As[0][r1][c1]);
    copy16_g2l(Bw + (size_t)(n0 + r1) * ldb + c1, &Bs[0][r1][c1]);
  }

  for (int k0 = 0; k0 < K; k0 += 32) {
    const int cur = (k0 >> 5) & 1;
    wait_async_all();      // this wave's copies for `cur` landed
    __syncthreads();       // everyone's landed; everyone done reading cur^1

    if (k0 + 32 < K) {     // overlap next tile's DMA with this tile's WMMAs
      const int nxt = cur ^ 1, kn = k0 + 32;
      copy16_g2l(A + (size_t)(m0 + r0) * lda + kn + c0, &As[nxt][r0][c0]);
      copy16_g2l(Bw + (size_t)(n0 + r0) * ldb + kn + c0, &Bs[nxt][r0][c0]);
      copy16_g2l(A + (size_t)(m0 + r1) * lda + kn + c1, &As[nxt][r1][c1]);
      copy16_g2l(Bw + (size_t)(n0 + r1) * ldb + kn + c1, &Bs[nxt][r1][c1]);
    }

    v16bf af[4];
#pragma unroll
    for (int mi = 0; mi < 4; ++mi)
      af[mi] = frag_a(&As[cur][wm * 64 + mi * 16][0], 40);
#pragma unroll
    for (int ni = 0; ni < 2; ++ni) {
      v16bf bfrag = frag_b(&Bs[cur][wn * 32 + ni * 16][0], 40);
#pragma unroll
      for (int mi = 0; mi < 4; ++mi)
        acc[mi][ni] = WMMA_BF16(af[mi], bfrag, acc[mi][ni]);
    }
  }

  // C layout: VGPR r -> row r + 8h, col = lane&15.
#pragma unroll
  for (int mi = 0; mi < 4; ++mi) {
#pragma unroll
    for (int ni = 0; ni < 2; ++ni) {
      const int rb = m0 + wm * 64 + mi * 16 + 8 * h;
      const int col = n0 + wn * 32 + ni * 16 + nlo;
#pragma unroll
      for (int r = 0; r < 8; ++r)
        C[(size_t)(rb + r) * ldc + col] = (OutT)acc[mi][ni][r];
    }
  }
}

// ---------------------------------------------------------------------------
// Flash attention: one block (128 thr = 4 waves) per (b, h, 64-row q tile).
// qkv layout: [B, S, 3D] bf16; q at col 0, k at D, v at 2D; head offset h*128.
// Each wave owns 16 q-rows: S strip 16x64 (4 tiles), O strip 16x128 (8 tiles).
// Q/K tiles staged via async global->LDS; V transposed into LDS on the fly.
// ---------------------------------------------------------------------------
__global__ void __launch_bounds__(128)
attn_flash(const bf16* __restrict__ qkv, bf16* __restrict__ attn) {
  __shared__ bf16 Qs[64][136];      // 64 x 128 + pad
  __shared__ bf16 Ks[64][136];
  __shared__ bf16 Vt[128][72];      // V transposed: Vt[d][kv_row]
  __shared__ bf16 Ps[4][16][72];    // per-wave P strip (16 x 64 + pad)

  const int tid = threadIdx.x, lane = tid & 31, wave = tid >> 5;
  const int qb = blockIdx.x, hh = blockIdx.y, bb = blockIdx.z;
  const int hl = lane >> 4, nlo = lane & 15;

  const bf16* gQ = qkv + (size_t)bb * Sn * (3 * Dn) + (size_t)hh * HDn;
  const bf16* gK = gQ + Dn;
  const bf16* gV = gQ + 2 * Dn;

  // Stage 64x128 Q tile (8 x 16B chunks per thread) — async into LDS.
#pragma unroll
  for (int it = 0; it < 8; ++it) {
    int idx = tid + it * 128;
    int r = idx >> 4, c = (idx & 15) << 3;
    copy16_g2l(gQ + (size_t)(qb * 64 + r) * (3 * Dn) + c, &Qs[r][c]);
  }

  v8f accO[8];
#pragma unroll
  for (int vt = 0; vt < 8; ++vt) accO[vt] = vzero8();
  float mrow[8], lrow[8];
#pragma unroll
  for (int r = 0; r < 8; ++r) { mrow[r] = -1e30f; lrow[r] = 0.0f; }

  for (int j = 0; j <= qb; ++j) {
    __syncthreads();  // protect Ks/Vt from previous iteration's readers
#pragma unroll
    for (int it = 0; it < 8; ++it) {
      int idx = tid + it * 128;
      int r = idx >> 4, c = (idx & 15) << 3;
      copy16_g2l(gK + (size_t)(j * 64 + r) * (3 * Dn) + c, &Ks[r][c]);
      v8bf vv = *(const v8bf*)(gV + (size_t)(j * 64 + r) * (3 * Dn) + c);
#pragma unroll
      for (int i = 0; i < 8; ++i) Vt[c + i][r] = vv[i];  // transpose into LDS
    }
    wait_async_all();   // Q (first iter) + K copies landed
    __syncthreads();

    // S = Q K^T  (16x64 strip per wave, K-dim 128 in 4 steps of 32)
    v8f accS[4];
#pragma unroll
    for (int t = 0; t < 4; ++t) accS[t] = vzero8();
#pragma unroll
    for (int kk = 0; kk < 4; ++kk) {
      v16bf aq = frag_a(&Qs[wave * 16][kk * 32], 136);
#pragma unroll
      for (int t = 0; t < 4; ++t) {
        v16bf bk = frag_b(&Ks[t * 16][kk * 32], 136);
        accS[t] = WMMA_BF16(aq, bk, accS[t]);
      }
    }

    // Scale + causal mask (only the diagonal k-tile needs element masking).
#pragma unroll
    for (int t = 0; t < 4; ++t) {
#pragma unroll
      for (int r = 0; r < 8; ++r) {
        float v = accS[t][r] * SCALE;
        if (j == qb) {
          int row_g = (qb << 6) + (wave << 4) + (hl << 3) + r;
          int col_g = (j << 6) + (t << 4) + nlo;
          if (col_g > row_g) v = -1e30f;
        }
        accS[t][r] = v;
      }
    }

    // Online softmax: per-row max/sum; rows live across 16-lane column groups.
    float mnew[8];
#pragma unroll
    for (int r = 0; r < 8; ++r) {
      float v = mrow[r];
#pragma unroll
      for (int t = 0; t < 4; ++t) v = fmaxf(v, accS[t][r]);
      v = fmaxf(v, __shfl_xor(v, 1));
      v = fmaxf(v, __shfl_xor(v, 2));
      v = fmaxf(v, __shfl_xor(v, 4));
      v = fmaxf(v, __shfl_xor(v, 8));
      mnew[r] = v;
    }
    float sf[8], rsum[8];
#pragma unroll
    for (int r = 0; r < 8; ++r) {
      sf[r] = __expf(mrow[r] - mnew[r]);
      mrow[r] = mnew[r];
      rsum[r] = 0.0f;
    }
#pragma unroll
    for (int t = 0; t < 4; ++t) {
#pragma unroll
      for (int r = 0; r < 8; ++r) {
        float p = __expf(accS[t][r] - mnew[r]);
        rsum[r] += p;
        Ps[wave][(hl << 3) + r][(t << 4) + nlo] = (bf16)p;  // C-layout -> LDS
      }
    }
#pragma unroll
    for (int r = 0; r < 8; ++r) {
      float v = rsum[r];
      v += __shfl_xor(v, 1);
      v += __shfl_xor(v, 2);
      v += __shfl_xor(v, 4);
      v += __shfl_xor(v, 8);
      lrow[r] = lrow[r] * sf[r] + v;
    }
#pragma unroll
    for (int vt = 0; vt < 8; ++vt)
#pragma unroll
      for (int r = 0; r < 8; ++r) accO[vt][r] *= sf[r];

    // Same-wave LDS RAW (Ps stores -> frag_a loads): make the wait explicit.
    asm volatile("s_wait_dscnt 0" ::: "memory");

    // O += P @ V   (P 16x64 as A-frags from LDS; V^T rows give B-frags)
#pragma unroll
    for (int kk = 0; kk < 2; ++kk) {
      v16bf ap = frag_a(&Ps[wave][0][kk * 32], 72);
#pragma unroll
      for (int vt = 0; vt < 8; ++vt) {
        v16bf bv = frag_b(&Vt[vt * 16][kk * 32], 72);
        accO[vt] = WMMA_BF16(ap, bv, accO[vt]);
      }
    }
  }

  // Normalize and write bf16 attention output [B, S, D].
  float inv[8];
#pragma unroll
  for (int r = 0; r < 8; ++r) inv[r] = 1.0f / lrow[r];
  bf16* gO = attn + (size_t)bb * Sn * Dn + (size_t)hh * HDn;
#pragma unroll
  for (int vt = 0; vt < 8; ++vt) {
#pragma unroll
    for (int r = 0; r < 8; ++r) {
      int row = qb * 64 + wave * 16 + (hl << 3) + r;
      gO[(size_t)row * Dn + (vt << 4) + nlo] = (bf16)(accO[vt][r] * inv[r]);
    }
  }
}

// ---------------------------------------------------------------------------
// Launch
// ---------------------------------------------------------------------------
extern "C" void kernel_launch(void* const* d_in, const int* in_sizes, int n_in,
                              void* d_out, int out_size, void* d_ws, size_t ws_size,
                              hipStream_t stream) {
  const float* x     = (const float*)d_in[0];
  const float* Wqkv  = (const float*)d_in[1];
  const float* Wproj = (const float*)d_in[2];
  float* out = (float*)d_out;

  const size_t nx = (size_t)Bn * Sn * Dn;        // 8.4M
  const size_t nq = (size_t)3 * Dn * Dn;         // 12.6M
  const size_t np = (size_t)Dn * Dn;             // 4.2M

  char* ws = (char*)d_ws;
  bf16* xb    = (bf16*)ws;  ws += nx * 2;
  bf16* wqkvb = (bf16*)ws;  ws += nq * 2;
  bf16* wprojb= (bf16*)ws;  ws += np * 2;
  bf16* qkvb  = (bf16*)ws;  ws += (size_t)Bn * Sn * 3 * Dn * 2;
  bf16* attnb = (bf16*)ws;  // nx * 2 bytes

  cvt_f32_bf16<<<2048, 256, 0, stream>>>(x, xb, (int)(nx / 4));
  cvt_f32_bf16<<<2048, 256, 0, stream>>>(Wqkv, wqkvb, (int)(nq / 4));
  cvt_f32_bf16<<<2048, 256, 0, stream>>>(Wproj, wprojb, (int)(np / 4));

  // qkv = x @ Wqkv^T : M=4096, N=6144, K=2048 -> bf16
  dim3 g1(4096 / 128, 6144 / 128);
  gemm_bf16_wmma<bf16><<<g1, 256, 0, stream>>>(xb, wqkvb, qkvb,
                                               4096, 6144, 2048,
                                               2048, 2048, 6144);

  // flash attention: grid (S/64, H, B)
  dim3 ga(Sn / 64, Hn, Bn);
  attn_flash<<<ga, 128, 0, stream>>>(qkvb, attnb);

  // out = attn @ Wproj^T : M=4096, N=2048, K=2048 -> f32
  dim3 g2(4096 / 128, 2048 / 128);
  gemm_bf16_wmma<float><<<g2, 256, 0, stream>>>(attnb, wprojb, out,
                                                4096, 2048, 2048,
                                                2048, 2048, 2048);
}
